// GraphAttentionLayer_21620865368429
// MI455X (gfx1250) — compile-verified, matrix operand order checked by reading, not run on hardware
//
#include <hip/hip_runtime.h>
#include <hip/hip_bf16.h>

typedef __attribute__((ext_vector_type(16))) _Float16 v16h;
typedef __attribute__((ext_vector_type(8)))  float    v8f;
typedef __attribute__((ext_vector_type(2)))  float    vf2;

#define N_NODES 8192
#define F_DIM   128
#define NEGTH   (-1e8f)

// ---------------------------------------------------------------------------
// Kernel 1: proj = x @ W[h]  (f32 accumulate), produce:
//   - s_src[h][n], s_tgt[h][n]  (f32)
//   - Vpk: proj in f16, pre-swizzled into WMMA B-fragment layout
//     Vpk[h][c][t][lane][half], c = j>>5 (K-chunk of 32), t = o>>4 (N-tile),
//     lane = ((K>>3)&1)<<4 | (o&15), half = (K&7) | (((K>>4)&1)<<3), K = j&31
// grid = H*N blocks, 128 threads
// ---------------------------------------------------------------------------
__global__ void k_proj(const float* __restrict__ x, const float* __restrict__ W,
                       const float* __restrict__ a_src, const float* __restrict__ a_tgt,
                       float* __restrict__ s_src, float* __restrict__ s_tgt,
                       _Float16* __restrict__ Vpk) {
    const int b = blockIdx.x;
    const int h = b >> 13;            // N = 8192 = 1<<13
    const int n = b & (N_NODES - 1);
    const int o = threadIdx.x;        // 0..127

    __shared__ float xl[F_DIM];
    __shared__ float red[F_DIM];

    xl[o] = x[(size_t)n * F_DIM + o];
    __syncthreads();

    const float* Wh = W + (size_t)h * F_DIM * F_DIM;
    float acc = 0.f;
#pragma unroll 8
    for (int k = 0; k < F_DIM; ++k)
        acc = fmaf(xl[k], Wh[(size_t)k * F_DIM + o], acc);

    // swizzled f16 V store (B-fragment order)
    {
        const int c = n >> 5, K = n & 31;
        const int lane = (((K >> 3) & 1) << 4) | (o & 15);
        const int half = (K & 7) | (((K >> 4) & 1) << 3);
        const int t = o >> 4;
        const size_t idx = ((((size_t)(h * (N_NODES / 32) + c)) * 8 + t) * 32 + lane) * 16 + half;
        Vpk[idx] = (_Float16)acc;
    }

    // s_src reduction
    red[o] = acc * a_src[h * F_DIM + o];
    __syncthreads();
    for (int s = 64; s > 0; s >>= 1) {
        if (o < s) red[o] += red[o + s];
        __syncthreads();
    }
    if (o == 0) s_src[h * N_NODES + n] = red[0];
    __syncthreads();

    // s_tgt reduction
    red[o] = acc * a_tgt[h * F_DIM + o];
    __syncthreads();
    for (int s = 64; s > 0; s >>= 1) {
        if (o < s) red[o] += red[o + s];
        __syncthreads();
    }
    if (o == 0) s_tgt[h * N_NODES + n] = red[0];
}

// ---------------------------------------------------------------------------
// merge of (max, sum-of-exp) pairs; d == 0 is the "empty" sentinel
// ---------------------------------------------------------------------------
__device__ __forceinline__ void softmax_merge(float& m, float& d, float m2, float d2) {
    if (d2 == 0.f) return;
    if (d == 0.f) { m = m2; d = d2; return; }
    if (m2 > m) { d = d * __expf(m - m2) + d2; m = m2; }
    else        { d += d2 * __expf(m2 - m); }
}

// ---------------------------------------------------------------------------
// Kernel 2a: partial column-softmax stats over 1024-row chunks.
// grid = 128 col-blocks * 8 row-chunks = 1024 blocks, 256 threads
// ---------------------------------------------------------------------------
__global__ void k_stats_partial(const float* __restrict__ mask,
                                const float* __restrict__ s_src,
                                const float* __restrict__ s_tgt,
                                float* __restrict__ pm, float* __restrict__ pd) {
    const int t = threadIdx.x;
    const int jl = t & 63, part = t >> 6;
    const int cb = blockIdx.x & 127;
    const int rc = blockIdx.x >> 7;
    const int j = cb * 64 + jl;

    const float st0 = s_tgt[j];
    const float st1 = s_tgt[N_NODES + j];
    float m0 = -INFINITY, d0 = 0.f, m1 = -INFINITY, d1 = 0.f;

    const int ibase = rc * 1024;
#pragma unroll 4
    for (int i = ibase + part; i < ibase + 1024; i += 4) {
        float mv = __builtin_nontemporal_load(mask + (size_t)i * N_NODES + j);
        if (mv > NEGTH) {               // unmasked => mask value is exactly 0
            float z0 = s_src[i] + st0;            z0 = z0 > 0.f ? z0 : 0.2f * z0;
            float z1 = s_src[N_NODES + i] + st1;  z1 = z1 > 0.f ? z1 : 0.2f * z1;
            if (z0 > m0) { d0 = d0 * __expf(m0 - z0) + 1.f; m0 = z0; } else d0 += __expf(z0 - m0);
            if (z1 > m1) { d1 = d1 * __expf(m1 - z1) + 1.f; m1 = z1; } else d1 += __expf(z1 - m1);
        }
    }

    __shared__ float rm[4][64][2];
    __shared__ float rd[4][64][2];
    rm[part][jl][0] = m0; rd[part][jl][0] = d0;
    rm[part][jl][1] = m1; rd[part][jl][1] = d1;
    __syncthreads();
    if (part == 0) {
        float m[2] = { m0, m1 }, d[2] = { d0, d1 };
#pragma unroll
        for (int p = 1; p < 4; ++p)
#pragma unroll
            for (int h = 0; h < 2; ++h)
                softmax_merge(m[h], d[h], rm[p][jl][h], rd[p][jl][h]);
#pragma unroll
        for (int h = 0; h < 2; ++h) {
            pm[(size_t)(h * 8 + rc) * N_NODES + j] = m[h];
            pd[(size_t)(h * 8 + rc) * N_NODES + j] = d[h];
        }
    }
}

// ---------------------------------------------------------------------------
// Kernel 2b: combine 8 partials -> packed column params
//   colp[h*N + j] = { s_tgt, Mcol, 1/D, 0 }  (one b128 load in kernel 3)
// grid = H*N/256 blocks, 256 threads
// ---------------------------------------------------------------------------
__global__ void k_stats_final(const float* __restrict__ pm, const float* __restrict__ pd,
                              const float* __restrict__ s_tgt,
                              float4* __restrict__ colp) {
    const int gid = blockIdx.x * 256 + threadIdx.x;   // h*N + j
    const int h = gid >> 13, j = gid & (N_NODES - 1);
    float m = -INFINITY, d = 0.f;
#pragma unroll
    for (int rc = 0; rc < 8; ++rc)
        softmax_merge(m, d, pm[(size_t)(h * 8 + rc) * N_NODES + j],
                            pd[(size_t)(h * 8 + rc) * N_NODES + j]);
    float rD = (d > 0.f) ? (1.f / d) : 0.f;
    colp[gid] = make_float4(s_tgt[gid], m, rD, 0.f);
}

// ---------------------------------------------------------------------------
// Kernel 3: out[i,:] = elu( 0.5*sum_h sum_j P[h,i,j]*proj[h,j,:] + 0.5*sum_h bias + x[i,:] )
// Workgroup = 16 output rows, 512 threads = 16 waves = {h:2} x {n-tile:8}.
// K=64 per step, two v_wmma_f32_16x16x32_f16 per wave per step; mask + B
// prefetch manually double-buffered (ping-pong registers, 2x unroll).
// grid = N/16 = 512 blocks, 512 threads
// ---------------------------------------------------------------------------
__global__ void __launch_bounds__(512)
k_attn_wmma(const float* __restrict__ mask, const float* __restrict__ x,
            const float* __restrict__ bias, const float* __restrict__ s_src,
            const float4* __restrict__ colp,
            const _Float16* __restrict__ Vpk, float* __restrict__ out) {
    const int tid  = threadIdx.x;
    const int i0   = blockIdx.x * 16;
    const int li   = tid >> 5;        // phase-1 role: local row 0..15
    const int jp   = tid & 31;        // phase-1 role: covers cols 2*jp, 2*jp+1 of 64
    const int wave = tid >> 5;        // phase-2 role
    const int lane = tid & 31;
    const int h    = wave >> 3;       // head
    const int t    = wave & 7;        // 16-wide N tile

    __shared__ __align__(32) _Float16 pA[2][2][32][16];  // [head][sub-chunk][lane][half]
    __shared__ float Cbuf[2][8][16][16];                 // epilogue staging (16KB)

    const float ss0 = s_src[i0 + li];
    const float ss1 = s_src[N_NODES + i0 + li];

    // A-fragment slots for this thread's two elements (same sub-chunk, K even/odd)
    const int sub    = jp >> 4;
    const int Keven  = (2 * jp) & 31;
    const int alane  = li | (((Keven >> 3) & 1) << 4);
    const int ahalf0 = (Keven & 7)       | (((Keven >> 4) & 1) << 3);
    const int ahalf1 = ((Keven + 1) & 7) | ((((Keven + 1) >> 4) & 1) << 3);

    v8f acc = { 0.f, 0.f, 0.f, 0.f, 0.f, 0.f, 0.f, 0.f };
    const v16h* vp   = (const v16h*)Vpk;
    const v16h* pAv0 = (const v16h*)(&pA[h][0][lane][0]);
    const v16h* pAv1 = (const v16h*)(&pA[h][1][lane][0]);

    const size_t bstr  = 8u * 32u;             // v16h per 32-chunk per head
    const size_t bbase = (size_t)h * (N_NODES / 32) * bstr + (size_t)t * 32 + lane;

    const vf2* mptr = (const vf2*)(mask + (size_t)(i0 + li) * N_NODES) + jp;

    auto step = [&](int c, vf2 mcur, v16h bc0, v16h bc1) {
        const int jg0 = c * 64 + 2 * jp;       // even column
        float p00 = 0.f, p01 = 0.f, p10 = 0.f, p11 = 0.f;
        if (mcur.x > NEGTH) {
            float4 c0 = colp[jg0];
            float4 c1 = colp[N_NODES + jg0];
            float z0 = ss0 + c0.x;  z0 = z0 > 0.f ? z0 : 0.2f * z0;
            float z1 = ss1 + c1.x;  z1 = z1 > 0.f ? z1 : 0.2f * z1;
            p00 = __expf(z0 - c0.y) * c0.z;
            p10 = __expf(z1 - c1.y) * c1.z;
        }
        if (mcur.y > NEGTH) {
            float4 c0 = colp[jg0 + 1];
            float4 c1 = colp[N_NODES + jg0 + 1];
            float z0 = ss0 + c0.x;  z0 = z0 > 0.f ? z0 : 0.2f * z0;
            float z1 = ss1 + c1.x;  z1 = z1 > 0.f ? z1 : 0.2f * z1;
            p01 = __expf(z0 - c0.y) * c0.z;
            p11 = __expf(z1 - c1.y) * c1.z;
        }
        __syncthreads();                       // prior A reads complete
        pA[0][sub][alane][ahalf0] = (_Float16)p00;
        pA[0][sub][alane][ahalf1] = (_Float16)p01;
        pA[1][sub][alane][ahalf0] = (_Float16)p10;
        pA[1][sub][alane][ahalf1] = (_Float16)p11;
        __syncthreads();

        v16h a0 = *pAv0;
        v16h a1 = *pAv1;
        acc = __builtin_amdgcn_wmma_f32_16x16x32_f16(
                  false, a0, false, bc0, (short)0, acc, false, false);
        acc = __builtin_amdgcn_wmma_f32_16x16x32_f16(
                  false, a1, false, bc1, (short)0, acc, false, false);
    };

    const int NITER = N_NODES / 64;            // 128 (even)
    vf2  mA  = __builtin_nontemporal_load(mptr);
    v16h bA0 = vp[bbase + 0 * bstr];
    v16h bA1 = vp[bbase + 1 * bstr];

    for (int c = 0; c < NITER; c += 2) {
        vf2  mB  = __builtin_nontemporal_load(mptr + (size_t)(c + 1) * 32);
        v16h bB0 = vp[bbase + (size_t)(2 * c + 2) * bstr];
        v16h bB1 = vp[bbase + (size_t)(2 * c + 3) * bstr];
        step(c, mA, bA0, bA1);

        const int cn = (c + 2 < NITER) ? (c + 2) : 0;   // uniform clamp on last trip
        mA  = __builtin_nontemporal_load(mptr + (size_t)cn * 32);
        bA0 = vp[bbase + (size_t)(2 * cn + 0) * bstr];
        bA1 = vp[bbase + (size_t)(2 * cn + 1) * bstr];
        step(c + 1, mB, bB0, bB1);
    }

    // ---- epilogue: head mean + bias + residual + elu ----
    __syncthreads();
#pragma unroll
    for (int r = 0; r < 8; ++r) {
        const int mrow = r + ((lane >> 4) << 3);   // C/D layout: VGPR r -> M=r / r+8
        Cbuf[h][t][mrow][lane & 15] = acc[r];
    }
    __syncthreads();

    for (int e = tid; e < 16 * F_DIM; e += 512) {
        const int i = e >> 7, o = e & 127;
        float v = 0.5f * (Cbuf[0][o >> 4][i][o & 15] + Cbuf[1][o >> 4][i][o & 15]);
        v += 0.5f * (bias[o] + bias[F_DIM + o]);
        v += x[(size_t)(i0 + i) * F_DIM + o];
        v = v > 0.f ? v : (__expf(v) - 1.f);       // elu, alpha=1
        out[(size_t)(i0 + i) * F_DIM + o] = v;
    }
}

// ---------------------------------------------------------------------------
// launch
// ---------------------------------------------------------------------------
extern "C" void kernel_launch(void* const* d_in, const int* in_sizes, int n_in,
                              void* d_out, int out_size, void* d_ws, size_t ws_size,
                              hipStream_t stream) {
    const float* x     = (const float*)d_in[0];
    const float* mask  = (const float*)d_in[1];
    const float* W     = (const float*)d_in[2];
    const float* a_src = (const float*)d_in[3];
    const float* a_tgt = (const float*)d_in[4];
    const float* bias  = (const float*)d_in[5];
    float* out = (float*)d_out;

    // workspace carve-up (~5.6 MB)
    char* ws = (char*)d_ws;
    float*    s_src = (float*)(ws);                 // H*N*4   = 65536
    float*    s_tgt = (float*)(ws + 65536);         // 65536
    float*    pm    = (float*)(ws + 131072);        // H*8*N*4 = 524288
    float*    pd    = (float*)(ws + 655360);        // 524288
    float4*   colp  = (float4*)(ws + 1179648);      // H*N*16  = 262144
    _Float16* Vpk   = (_Float16*)(ws + 1441792);    // H*N*128*2 = 4194304

    k_proj<<<2 * N_NODES, 128, 0, stream>>>(x, W, a_src, a_tgt, s_src, s_tgt, Vpk);
    k_stats_partial<<<1024, 256, 0, stream>>>(mask, s_src, s_tgt, pm, pd);
    k_stats_final<<<(2 * N_NODES) / 256, 256, 0, stream>>>(pm, pd, s_tgt, colp);
    k_attn_wmma<<<N_NODES / 16, 512, 0, stream>>>(mask, x, bias, s_src, colp, Vpk, out);

    // second tuple element: the (unchanged) mask, concatenated after out
    hipMemcpyAsync(out + (size_t)N_NODES * F_DIM, mask,
                   (size_t)N_NODES * N_NODES * sizeof(float),
                   hipMemcpyDeviceToDevice, stream);
}
